// CML_Attiention_807453852215
// MI455X (gfx1250) — compile-verified, matrix operand order checked by reading, not run on hardware
//
#include <hip/hip_runtime.h>
#include <hip/hip_bf16.h>
#include <stdint.h>

// ---------------------------------------------------------------------------
// CDNA5 / gfx1250 implementation of the CML cross-modal fusion block.
// Dense GEMMs: v_wmma_f32_16x16x32_bf16, register-blocked 32x64 per wave
// (2x4 tiles of 16x16) for 2.7x higher FLOP/byte vs one-tile-per-wave.
// ---------------------------------------------------------------------------

typedef __attribute__((ext_vector_type(16))) __bf16 v16bf;
typedef __attribute__((ext_vector_type(8)))  float  v8f;

#define DIMC   512
#define NHEADS 16
#define HD     32
#define BB     4
#define NTOK   4096
#define NQ     2048

static __device__ __forceinline__ unsigned short f2bf(float f) {
    unsigned int u = __float_as_uint(f);
    unsigned int r = (u + 0x7FFFu + ((u >> 16) & 1u)) >> 16;   // RNE
    return (unsigned short)r;
}
static __device__ __forceinline__ float bf2f(unsigned short h) {
    return __uint_as_float(((unsigned int)h) << 16);
}

// ---------------------------------------------------------------------------
// Generic bf16 GEMM:  out = act(A[MxK] * Wt^T + bias)  (Wt stored N x K)
// mode 0: bf16 out, 1: fp32 out, 2: fp32 out + residual
// act  0: none, 1: gelu(exact erf), 2: relu
// Each wave owns a 32x64 block = 2(M) x 4(N) WMMA tiles; block = 8 waves.
// Requires M % 32 == 0, N % 64 == 0, K % 32 == 0.
// ---------------------------------------------------------------------------
__global__ void gemm_bf16_wmma(const unsigned short* __restrict__ A,
                               const unsigned short* __restrict__ Wt,
                               const float* __restrict__ bias,
                               unsigned short* __restrict__ outb,
                               float* __restrict__ outf,
                               const float* __restrict__ residual,
                               int M, int N, int K, int act, int mode) {
    const int wave = threadIdx.x >> 5;
    const int lane = threadIdx.x & 31;
    const int tn = N >> 6;                          // 64-wide N blocks
    const int tm = M >> 5;                          // 32-tall M blocks
    int blk = blockIdx.x * 8 + wave;
    if (blk >= tn * tm) return;                     // wave-uniform exit
    const int blk_n = blk % tn;
    const int blk_m = blk / tn;

    const int r  = lane & 15;                       // row within fragment
    const int kb = (lane >> 4) * 8;                 // K sub-block per lane half

    const unsigned short* arow0 = A  + (size_t)(blk_m * 32 + r)      * K + kb;
    const unsigned short* arow1 = A  + (size_t)(blk_m * 32 + 16 + r) * K + kb;
    const unsigned short* brow0 = Wt + (size_t)(blk_n * 64 + r)      * K + kb;
    const unsigned short* brow1 = Wt + (size_t)(blk_n * 64 + 16 + r) * K + kb;
    const unsigned short* brow2 = Wt + (size_t)(blk_n * 64 + 32 + r) * K + kb;
    const unsigned short* brow3 = Wt + (size_t)(blk_n * 64 + 48 + r) * K + kb;

    v8f c00 = {}, c01 = {}, c02 = {}, c03 = {};
    v8f c10 = {}, c11 = {}, c12 = {}, c13 = {};

    union Frag { uint4 q[2]; v16bf v; };
    for (int k0 = 0; k0 < K; k0 += 32) {
        Frag fa0, fa1, fb0, fb1, fb2, fb3;
        fa0.q[0] = *(const uint4*)(arow0 + k0);
        fa0.q[1] = *(const uint4*)(arow0 + k0 + 16);
        fa1.q[0] = *(const uint4*)(arow1 + k0);
        fa1.q[1] = *(const uint4*)(arow1 + k0 + 16);
        fb0.q[0] = *(const uint4*)(brow0 + k0);
        fb0.q[1] = *(const uint4*)(brow0 + k0 + 16);
        fb1.q[0] = *(const uint4*)(brow1 + k0);
        fb1.q[1] = *(const uint4*)(brow1 + k0 + 16);
        fb2.q[0] = *(const uint4*)(brow2 + k0);
        fb2.q[1] = *(const uint4*)(brow2 + k0 + 16);
        fb3.q[0] = *(const uint4*)(brow3 + k0);
        fb3.q[1] = *(const uint4*)(brow3 + k0 + 16);
        c00 = __builtin_amdgcn_wmma_f32_16x16x32_bf16(false, fa0.v, false, fb0.v, (short)0, c00, false, false);
        c01 = __builtin_amdgcn_wmma_f32_16x16x32_bf16(false, fa0.v, false, fb1.v, (short)0, c01, false, false);
        c02 = __builtin_amdgcn_wmma_f32_16x16x32_bf16(false, fa0.v, false, fb2.v, (short)0, c02, false, false);
        c03 = __builtin_amdgcn_wmma_f32_16x16x32_bf16(false, fa0.v, false, fb3.v, (short)0, c03, false, false);
        c10 = __builtin_amdgcn_wmma_f32_16x16x32_bf16(false, fa1.v, false, fb0.v, (short)0, c10, false, false);
        c11 = __builtin_amdgcn_wmma_f32_16x16x32_bf16(false, fa1.v, false, fb1.v, (short)0, c11, false, false);
        c12 = __builtin_amdgcn_wmma_f32_16x16x32_bf16(false, fa1.v, false, fb2.v, (short)0, c12, false, false);
        c13 = __builtin_amdgcn_wmma_f32_16x16x32_bf16(false, fa1.v, false, fb3.v, (short)0, c13, false, false);
    }

    // epilogue: 8 tiles
    v8f* accs[2][4] = {{&c00, &c01, &c02, &c03}, {&c10, &c11, &c12, &c13}};
#pragma unroll
    for (int i = 0; i < 2; ++i) {
        const int mbase = blk_m * 32 + i * 16 + ((lane < 16) ? 0 : 8);
#pragma unroll
        for (int j = 0; j < 4; ++j) {
            const int n = blk_n * 64 + j * 16 + r;
            const float bs = bias ? bias[n] : 0.f;
            v8f& c = *accs[i][j];
#pragma unroll
            for (int v = 0; v < 8; ++v) {
                float x = c[v] + bs;
                if (act == 1)      x = 0.5f * x * (1.f + erff(x * 0.70710678118654752f));
                else if (act == 2) x = fmaxf(x, 0.f);
                size_t o = (size_t)(mbase + v) * N + n;
                if (mode == 0)      outb[o] = f2bf(x);
                else if (mode == 1) outf[o] = x;
                else                outf[o] = x + residual[o];
            }
        }
    }
}

// LN over channel dim; input NCHW fp32, output (B*N) x C bf16 row-major
__global__ void ln1_kernel(const float* __restrict__ in,
                           const float* __restrict__ g, const float* __restrict__ b2,
                           unsigned short* __restrict__ out) {
    const int tok = blockIdx.x;            // 0..B*NTOK-1
    const int b = tok / NTOK, n = tok % NTOK;
    const int c = threadIdx.x;             // 512 threads
    float x = in[((size_t)b * DIMC + c) * NTOK + n];
    __shared__ float s1[DIMC], s2[DIMC];
    s1[c] = x; s2[c] = x * x;
    __syncthreads();
    for (int st = DIMC / 2; st > 0; st >>= 1) {
        if (c < st) { s1[c] += s1[c + st]; s2[c] += s2[c + st]; }
        __syncthreads();
    }
    float mean = s1[0] * (1.f / DIMC);
    float var  = s2[0] * (1.f / DIMC) - mean * mean;
    float inv  = rsqrtf(var + 1e-5f);
    out[(size_t)tok * DIMC + c] = f2bf((x - mean) * inv * g[c] + b2[c]);
}

// LN over rows of fp32 (B*NQ) x C -> bf16
__global__ void ln2_kernel(const float* __restrict__ in,
                           const float* __restrict__ g, const float* __restrict__ b2,
                           unsigned short* __restrict__ out) {
    const int row = blockIdx.x;
    const int c = threadIdx.x;
    float x = in[(size_t)row * DIMC + c];
    __shared__ float s1[DIMC], s2[DIMC];
    s1[c] = x; s2[c] = x * x;
    __syncthreads();
    for (int st = DIMC / 2; st > 0; st >>= 1) {
        if (c < st) { s1[c] += s1[c + st]; s2[c] += s2[c + st]; }
        __syncthreads();
    }
    float mean = s1[0] * (1.f / DIMC);
    float var  = s2[0] * (1.f / DIMC) - mean * mean;
    float inv  = rsqrtf(var + 1e-5f);
    out[(size_t)row * DIMC + c] = f2bf((x - mean) * inv * g[c] + b2[c]);
}

// fp32 K x N weight -> bf16 N x K (transposed)
__global__ void wprep_kernel(const float* __restrict__ W,
                             unsigned short* __restrict__ Wt, int K, int Nn) {
    int i = blockIdx.x * blockDim.x + threadIdx.x;
    if (i >= K * Nn) return;
    int k = i / Nn, n = i % Nn;
    Wt[(size_t)n * K + k] = f2bf(W[i]);
}

// aq[row] = concat(xa[idx], xb[idx])  (bf16 rows of 1024)
__global__ void gather_kernel(const unsigned short* __restrict__ xa,
                              const unsigned short* __restrict__ xb,
                              const int* __restrict__ idx,
                              unsigned short* __restrict__ aq) {
    const int row = blockIdx.x;                 // B*NQ
    const int b = row >> 11;
    const int t = idx[row];
    const unsigned int* pa = (const unsigned int*)(xa + ((size_t)b * NTOK + t) * DIMC);
    const unsigned int* pb = (const unsigned int*)(xb + ((size_t)b * NTOK + t) * DIMC);
    unsigned int* po = (unsigned int*)(aq + (size_t)row * 2 * DIMC);
    const int tid = threadIdx.x;                // 256 -> 256 dwords each half
    po[tid]       = pa[tid];
    po[256 + tid] = pb[tid];
}

// 3x3 neighborhood attention; one wave per head (wave32), block = 16 waves
__global__ void sna_attn_kernel(const unsigned short* __restrict__ qbuf,   // (B*NQ) x 512
                                const unsigned short* __restrict__ kvbuf,  // (B*NTOK) x 1024
                                const float* __restrict__ rpb,             // 16 x 9
                                const int* __restrict__ idx,
                                unsigned short* __restrict__ attnout) {
    const int row = blockIdx.x;
    const int b = row >> 11;
    const int t = idx[row];
    const int h = threadIdx.x >> 5;
    const int d = threadIdx.x & 31;
    const int y = t >> 6, x = t & 63;
    const float q = bf2f(qbuf[(size_t)row * DIMC + h * HD + d]);
    const float scale = 0.17677669529663687f;   // 32^-0.5
    float scores[9], vv[9];
#pragma unroll
    for (int k = 0; k < 9; ++k) {
        int yy = y + k / 3 - 1, xx = x + k % 3 - 1;
        float kf = 0.f, vf = 0.f;
        if (yy >= 0 && yy < 64 && xx >= 0 && xx < 64) {
            size_t base = ((size_t)b * NTOK + yy * 64 + xx) * (2 * DIMC) + h * HD + d;
            kf = bf2f(kvbuf[base]);
            vf = bf2f(kvbuf[base + DIMC]);
        }
        float s = q * kf;
#pragma unroll
        for (int off = 16; off; off >>= 1) s += __shfl_xor(s, off, 32);
        scores[k] = -s * scale + rpb[h * 9 + k];
        vv[k] = vf;
    }
    float m = scores[0];
#pragma unroll
    for (int k = 1; k < 9; ++k) m = fmaxf(m, scores[k]);
    float den = 0.f, o = 0.f;
#pragma unroll
    for (int k = 0; k < 9; ++k) {
        float e = __expf(scores[k] - m);
        den += e; o += e * vv[k];
    }
    attnout[(size_t)row * DIMC + h * HD + d] = f2bf(o / den);
}

// gate = sigmoid(grelu @ g2_w + g2_b)   (N=2, done with block reduction)
__global__ void gate_kernel(const unsigned short* __restrict__ grelu,
                            const float* __restrict__ g2w,  // 512 x 2
                            const float* __restrict__ g2b,
                            float* __restrict__ gate) {
    const int row = blockIdx.x;
    const int tid = threadIdx.x;                // 256
    float a0 = 0.f, a1 = 0.f;
    for (int c = tid; c < DIMC; c += 256) {
        float xv = bf2f(grelu[(size_t)row * DIMC + c]);
        a0 += xv * g2w[c * 2 + 0];
        a1 += xv * g2w[c * 2 + 1];
    }
    __shared__ float s0[256], s1[256];
    s0[tid] = a0; s1[tid] = a1;
    __syncthreads();
    for (int st = 128; st > 0; st >>= 1) {
        if (tid < st) { s0[tid] += s0[tid + st]; s1[tid] += s1[tid + st]; }
        __syncthreads();
    }
    if (tid < 2) {
        float v = (tid == 0 ? s0[0] : s1[0]) + g2b[tid];
        gate[row * 2 + tid] = 1.f / (1.f + __expf(-v));
    }
}

// fused = g0*projout + g1*res  (res gathered from original NCHW input)
__global__ void fuse_kernel(const float* __restrict__ projout,
                            const float* __restrict__ gate,
                            const float* __restrict__ in_orig,
                            const int* __restrict__ idx,
                            float* __restrict__ fused) {
    const int row = blockIdx.x;
    const int c = threadIdx.x;                  // 512
    const int b = row >> 11;
    const int t = idx[row];
    float g0 = gate[row * 2], g1 = gate[row * 2 + 1];
    float res = in_orig[((size_t)b * DIMC + c) * NTOK + t];
    fused[(size_t)row * DIMC + c] = g0 * projout[(size_t)row * DIMC + c] + g1 * res;
}

__global__ void scatter_kernel(const float* __restrict__ fin,
                               const int* __restrict__ idx,
                               float* __restrict__ outh) {
    const int row = blockIdx.x;
    const int c = threadIdx.x;                  // 512
    const int b = row >> 11;
    const int t = idx[row];
    outh[((size_t)b * DIMC + c) * NTOK + t] = fin[(size_t)row * DIMC + c];
}

__global__ void fill_zero(float* p, size_t n) {
    size_t i = (size_t)blockIdx.x * blockDim.x + threadIdx.x;
    if (i < n) p[i] = 0.f;
}

// ---------------------------------------------------------------------------
extern "C" void kernel_launch(void* const* d_in, const int* in_sizes, int n_in,
                              void* d_out, int out_size, void* d_ws, size_t ws_size,
                              hipStream_t stream) {
    (void)in_sizes; (void)n_in; (void)out_size; (void)ws_size;
    const float* in_rgb = (const float*)d_in[0];
    const float* in_th  = (const float*)d_in[1];
    const int*   idx    = (const int*)d_in[2];

    // param index bases: sna_rgb=3..13, sna_th=14..24
    auto P = [&](int i) { return (const float*)d_in[i]; };

    const int MR = BB * NQ;        // 8192 gathered rows
    const int MF = BB * NTOK;      // 16384 full rows

    // workspace layout
    char* ws = (char*)d_ws;
    size_t off = 0;
    auto alloc = [&](size_t bytes) { size_t o = off; off += (bytes + 255) & ~(size_t)255; return o; };
    unsigned short* xn_rgb = (unsigned short*)(ws + alloc((size_t)MF * DIMC * 2));
    unsigned short* xn_th  = (unsigned short*)(ws + alloc((size_t)MF * DIMC * 2));
    unsigned short* wt_kv[2], *wt_q[2], *wt_proj[2], *wt_g1[2], *wt_fc1[2], *wt_fc2[2];
    for (int i = 0; i < 2; ++i) {
        wt_kv[i]   = (unsigned short*)(ws + alloc((size_t)1024 * 512 * 2));
        wt_q[i]    = (unsigned short*)(ws + alloc((size_t)512 * 1024 * 2));
        wt_proj[i] = (unsigned short*)(ws + alloc((size_t)512 * 512 * 2));
        wt_g1[i]   = (unsigned short*)(ws + alloc((size_t)512 * 1024 * 2));
        wt_fc1[i]  = (unsigned short*)(ws + alloc((size_t)1024 * 512 * 2));
        wt_fc2[i]  = (unsigned short*)(ws + alloc((size_t)512 * 1024 * 2));
    }
    unsigned short* kvbuf   = (unsigned short*)(ws + alloc((size_t)MF * 1024 * 2));  // also hidden
    unsigned short* aq      = (unsigned short*)(ws + alloc((size_t)MR * 1024 * 2));  // also final(fp32)
    unsigned short* qbuf    = (unsigned short*)(ws + alloc((size_t)MR * DIMC * 2));  // also ln2
    unsigned short* attnout = (unsigned short*)(ws + alloc((size_t)MR * DIMC * 2));
    float* projout = (float*)(ws + alloc((size_t)MR * DIMC * 4));
    unsigned short* grelu = (unsigned short*)(ws + alloc((size_t)MR * DIMC * 2));
    float* gate  = (float*)(ws + alloc((size_t)MR * 2 * 4));
    float* fused = (float*)(ws + alloc((size_t)MR * DIMC * 4));
    unsigned short* hidden = kvbuf;        // reuse (dead after attention)
    unsigned short* ln2b   = qbuf;         // reuse (dead after attention)
    float* finalb = (float*)aq;            // reuse (aq dead after g1 GEMM)

    // zero outputs
    {
        size_t n = 2ull * BB * DIMC * NTOK;
        fill_zero<<<dim3((unsigned)((n + 255) / 256)), dim3(256), 0, stream>>>((float*)d_out, n);
    }
    // layernorm1 -> bf16 token-major activations
    ln1_kernel<<<dim3(MF), dim3(DIMC), 0, stream>>>(in_rgb, P(25), P(26), xn_rgb);
    ln1_kernel<<<dim3(MF), dim3(DIMC), 0, stream>>>(in_th,  P(27), P(28), xn_th);

    // weight prep (bf16, transposed to N x K)
    auto prep = [&](const float* w, unsigned short* wt, int K, int Nn) {
        int tot = K * Nn;
        wprep_kernel<<<dim3((tot + 255) / 256), dim3(256), 0, stream>>>(w, wt, K, Nn);
    };
    for (int br = 0; br < 2; ++br) {
        int pb = 3 + br * 11;
        prep(P(pb + 0), wt_kv[br],   512, 1024);
        prep(P(pb + 2), wt_q[br],    1024, 512);
        prep(P(pb + 4), wt_proj[br], 512, 512);
        prep(P(pb + 6), wt_g1[br],   1024, 512);
        prep(P(33 + br * 2), wt_fc1[br], 512, 1024);
        prep(P(34 + br * 2), wt_fc2[br], 1024, 512);
    }

    auto gemm = [&](const unsigned short* A, const unsigned short* Wt, const float* bias,
                    unsigned short* ob, float* of, const float* res,
                    int M, int N, int K, int act, int mode) {
        int blks = (M / 32) * (N / 64);            // 32x64 per wave
        gemm_bf16_wmma<<<dim3((blks + 7) / 8), dim3(256), 0, stream>>>(
            A, Wt, bias, ob, of, res, M, N, K, act, mode);
    };

    for (int br = 0; br < 2; ++br) {
        int pb = 3 + br * 11;
        const unsigned short* xn_x = (br == 0) ? xn_th  : xn_rgb;   // KV source
        const unsigned short* xn_y = (br == 0) ? xn_rgb : xn_th;    // query source
        const float* in_orig = (br == 0) ? in_rgb : in_th;
        float* out_half = (float*)d_out + (size_t)br * BB * DIMC * NTOK;

        // KV = xn_x @ kv_w + kv_b   (cols 0..511 = K, 512..1023 = V)
        gemm(xn_x, wt_kv[br], P(pb + 1), kvbuf, nullptr, nullptr, MF, 1024, 512, 0, 0);
        // gathered y_cat rows = [xn_x | xn_y]
        gather_kernel<<<dim3(MR), dim3(256), 0, stream>>>(xn_x, xn_y, idx, aq);
        // Q (only gathered rows)
        gemm(aq, wt_q[br], P(pb + 3), qbuf, nullptr, nullptr, MR, 512, 1024, 0, 0);
        // 3x3 neighborhood attention with RPB
        sna_attn_kernel<<<dim3(MR), dim3(512), 0, stream>>>(qbuf, kvbuf, P(pb + 10), idx, attnout);
        // projection
        gemm(attnout, wt_proj[br], P(pb + 5), nullptr, projout, nullptr, MR, 512, 512, 0, 1);
        // gate path: relu(y_cat @ g1) then tiny N=2 matvec + sigmoid
        gemm(aq, wt_g1[br], P(pb + 7), grelu, nullptr, nullptr, MR, 512, 1024, 2, 0);
        gate_kernel<<<dim3(MR), dim3(256), 0, stream>>>(grelu, P(pb + 8), P(pb + 9), gate);
        // fused = g0*proj + g1*residual(original input)
        fuse_kernel<<<dim3(MR), dim3(512), 0, stream>>>(projout, gate, in_orig, idx, fused);
        // MLP block: LN2, gelu fc1, fc2 + residual
        ln2_kernel<<<dim3(MR), dim3(DIMC), 0, stream>>>(fused, P(29 + br * 2), P(30 + br * 2), ln2b);
        gemm(ln2b, wt_fc1[br], nullptr, hidden, nullptr, nullptr, MR, 1024, 512, 1, 0);
        gemm(hidden, wt_fc2[br], nullptr, nullptr, finalb, fused, MR, 512, 1024, 0, 2);
        // scatter to NCHW output half
        scatter_kernel<<<dim3(MR), dim3(512), 0, stream>>>(finalb, idx, out_half);
    }
}
